// Transformer_68264210202659
// MI455X (gfx1250) — matrix operand order repaired and checked by reading, not compiled, not run on hardware
//
#include <hip/hip_runtime.h>
#include <hip/hip_bf16.h>
#include <math.h>

// Problem constants (from the reference)
#define BB   2
#define LLN  256
#define DDM  512
#define HH   8
#define DHH  64
#define SSP  9
#define RRB  64
#define FFE  (SSP * RRB)   // 576
#define DFF  2048

typedef __attribute__((ext_vector_type(16))) _Float16 v16h;
typedef __attribute__((ext_vector_type(8)))  _Float16 v8h;
typedef __attribute__((ext_vector_type(8)))  float    v8f;
typedef __attribute__((ext_vector_type(4)))  unsigned u32x4;
typedef __attribute__((ext_vector_type(4)))  int      i32x4;
typedef __attribute__((ext_vector_type(8)))  int      i32x8;

__device__ __forceinline__ v8f wmma_f16(v16h a, v16h b, v8f c) {
  // D = A(16x32 f16) x B(32x16 f16) + C(16x16 f32)
  return __builtin_amdgcn_wmma_f32_16x16x32_f16(
      /*neg_a=*/false, a, /*neg_b=*/false, b,
      /*c_mod=*/(short)0, c, /*reuse_a=*/false, /*reuse_b=*/false);
}

// ---------------------------------------------------------------------------
// Kernel 1: t[bl,h,f] = sum_d rp_w[f, h*64+d] * q[bl, h*64+d]     (151M MACs)
// ---------------------------------------------------------------------------
__global__ void k_rpproj_t(const float* __restrict__ src,
                           const float* __restrict__ rp_w,
                           float* __restrict__ t) {
  __shared__ float qs[DDM];
  const int bl = blockIdx.x, tid = threadIdx.x;
  const float* q = src + (size_t)bl * DDM;
  qs[tid]       = q[tid];
  qs[tid + 256] = q[tid + 256];
  __syncthreads();
  #pragma unroll 1
  for (int i = 0; i < (HH * FFE) / 256; ++i) {   // 18 outputs/thread
    const int idx = tid + i * 256;
    const int h = idx / FFE, f = idx % FFE;
    const float* w  = rp_w + (size_t)f * DDM + h * DHH;
    const float* qq = qs + h * DHH;
    float acc = 0.f;
    #pragma unroll
    for (int d = 0; d < DHH; ++d) acc = fmaf(qq[d], w[d], acc);
    t[(size_t)bl * HH * FFE + idx] = acc;
  }
}

// ---------------------------------------------------------------------------
// Kernel 2 (WMMA): scores[b,h,l,m] = (q_h[l,:]·q_h[m,:]) / 8
// ---------------------------------------------------------------------------
__global__ void k_base_scores(const float* __restrict__ src,
                              float* __restrict__ scores) {
  const int l0 = blockIdx.x * 16, m0 = blockIdx.y * 16;
  const int bh = blockIdx.z, b = bh / HH, h = bh % HH;
  const int lane = threadIdx.x, mr = lane & 15, half = lane >> 4;
  const float* q = src + ((size_t)b * LLN) * DDM + h * DHH;
  v8f c = {0.f, 0.f, 0.f, 0.f, 0.f, 0.f, 0.f, 0.f};
  #pragma unroll
  for (int kk = 0; kk < DHH; kk += 32) {
    v16h a, bm;
    #pragma unroll
    for (int i = 0; i < 16; ++i) {
      const int ka = kk + (half << 4) + i;
      a[i]  = (_Float16)q[(size_t)(l0 + mr) * DDM + ka];
      const int kb = kk + (half << 4) + i;
      bm[i] = (_Float16)q[(size_t)(m0 + mr) * DDM + kb];   // B = q^T
    }
    c = wmma_f16(a, bm, c);
  }
  float* out = scores + ((size_t)bh * LLN) * LLN;
  #pragma unroll
  for (int i = 0; i < 8; ++i)
    out[(size_t)(l0 + i + 8 * half) * LLN + m0 + mr] = c[i] * 0.125f;
}

// ---------------------------------------------------------------------------
// Kernel 3: scores[b,h,l,m] += sum_{s,r} sph[s]*rbf[r]*t[b,l,h,s*64+r]
// t row (18 KB) staged to LDS by the Tensor Data Mover (one DMA per block)
// ---------------------------------------------------------------------------
__global__ void k_rp_scores(const float* __restrict__ rel_diss,
                            const float* __restrict__ rel_dirs,
                            const float* __restrict__ t,
                            float* __restrict__ scores) {
  __shared__ float tl[HH * FFE];                 // 4608 floats = 18 KB
  const int bl = blockIdx.x, b = bl / LLN, l = bl % LLN;
  const int m = threadIdx.x;
  const float* gsrc = t + (size_t)bl * HH * FFE;

#if __has_builtin(__builtin_amdgcn_tensor_load_to_lds)
  if (threadIdx.x < 32) {   // wave 0 issues the TDM descriptor (EXEC ignored)
    const unsigned lds_off = (unsigned)(size_t)(void*)tl;   // addr[31:0] = LDS offset
    const unsigned long long ga = (unsigned long long)(const void*)gsrc;
    u32x4 g0 = {};
    g0[0] = 1u;                                           // count=1 (valid), user mode
    g0[1] = lds_off;                                      // lds_addr
    g0[2] = (unsigned)ga;                                 // global_addr[31:0]
    g0[3] = ((unsigned)(ga >> 32) & 0x01FFFFFFu) | (2u << 30);  // addr[56:32] | type=2
    i32x8 g1 = {};
    g1[0] = 0x00020000;                                   // wg_mask=0, data_size=4B
    g1[1] = (int)(((unsigned)(HH * FFE) & 0xFFFFu) << 16);// tensor_dim0[15:0]
    g1[2] = (int)(((unsigned)(HH * FFE) >> 16) | (1u << 16)); // dim0[31:16] | tensor_dim1=1
    g1[3] = (int)(((unsigned)(HH * FFE) & 0xFFFFu) << 16);// tile_dim0 = 4608
    g1[4] = 1;                                            // tile_dim1=1, tile_dim2=0
    g1[5] = HH * FFE;                                     // tensor_dim0_stride[31:0]
    g1[6] = (int)(((unsigned)(HH * FFE) & 0xFFFFu) << 16);// stride0[47:32]=0 | stride1[15:0]
    g1[7] = 0;                                            // stride1[47:16]=0
    const i32x4 z4 = {0, 0, 0, 0};
  #if __clang_major__ >= 23
    const i32x8 z8 = {0, 0, 0, 0, 0, 0, 0, 0};
    __builtin_amdgcn_tensor_load_to_lds(g0, g1, z4, z4, z8, 0);
  #else
    __builtin_amdgcn_tensor_load_to_lds(g0, g1, z4, z4, 0);
  #endif
    __builtin_amdgcn_s_wait_tensorcnt(0);
  }
  __syncthreads();
#else
  for (int i = m; i < HH * FFE; i += 256) tl[i] = gsrc[i];
  __syncthreads();
#endif

  const float d = rel_diss[(size_t)bl * LLN + m];
  const float* dir = rel_dirs + ((size_t)bl * LLN + m) * 3;
  const float x = dir[0], y = dir[1], z = dir[2];
  const float c0 = 0.28209479177387814f, c1 = 0.4886025119029199f;
  const float c2 = 1.0925484305920792f, c20 = 0.31539156525252005f,
              c22 = 0.5462742152960396f;
  const float sph[SSP] = {c0, c1 * y, c1 * z, c1 * x, c2 * x * y, c2 * y * z,
                          c20 * (3.f * z * z - 1.f), c2 * x * z,
                          c22 * (x * x - y * y)};
  float u = fminf(fmaxf(d * 0.1f, 0.f), 1.f);
  const float env = 0.5f * (__cosf(3.14159265358979323846f * u) + 1.f);
  const float step = 10.0f / 63.0f;
  const float w = 10.0f / 64.0f;
  const float inv2w2 = 1.0f / (2.0f * w * w);
  float rbf[RRB];
  #pragma unroll
  for (int r = 0; r < RRB; ++r) {
    const float dd = d - (float)r * step;
    rbf[r] = __expf(-dd * dd * inv2w2) * env;
  }
  #pragma unroll 1
  for (int h = 0; h < HH; ++h) {
    float acc = 0.f;
    #pragma unroll 1
    for (int s = 0; s < SSP; ++s) {
      const float* tp = tl + h * FFE + s * RRB;
      float p = 0.f;
      #pragma unroll
      for (int r = 0; r < RRB; ++r) p = fmaf(rbf[r], tp[r], p);
      acc = fmaf(sph[s], p, acc);
    }
    scores[(((size_t)(b * HH + h)) * LLN + l) * LLN + m] += acc;
  }
}

// ---------------------------------------------------------------------------
// Kernel 4: softmax over m (row length 256), one block per (b,h,l)
// ---------------------------------------------------------------------------
__global__ void k_softmax(float* __restrict__ scores) {
  __shared__ float red[256];
  const int row = blockIdx.x, tid = threadIdx.x;
  float* p = scores + (size_t)row * LLN;
  const float v = p[tid];
  red[tid] = v; __syncthreads();
  for (int s = 128; s > 0; s >>= 1) {
    if (tid < s) red[tid] = fmaxf(red[tid], red[tid + s]);
    __syncthreads();
  }
  const float mx = red[0]; __syncthreads();
  const float e = __expf(v - mx);
  red[tid] = e; __syncthreads();
  for (int s = 128; s > 0; s >>= 1) {
    if (tid < s) red[tid] += red[tid + s];
    __syncthreads();
  }
  p[tid] = e / red[0];
}

// ---------------------------------------------------------------------------
// Kernel 5 (WMMA): attn[b,l,h,d] = sum_m attn_w[b,h,l,m] * q[b,m,h,d]
// ---------------------------------------------------------------------------
__global__ void k_attn_v(const float* __restrict__ scores,
                         const float* __restrict__ src,
                         float* __restrict__ attn) {
  const int l0 = blockIdx.x * 16, n0 = blockIdx.y * 16;
  const int bh = blockIdx.z, b = bh / HH, h = bh % HH;
  const int lane = threadIdx.x, mr = lane & 15, half = lane >> 4;
  const float* aw = scores + ((size_t)bh * LLN) * LLN;
  const float* q = src + ((size_t)b * LLN) * DDM + h * DHH;
  v8f c = {0.f, 0.f, 0.f, 0.f, 0.f, 0.f, 0.f, 0.f};
  #pragma unroll 1
  for (int kk = 0; kk < LLN; kk += 32) {
    v16h a, bm;
    #pragma unroll
    for (int i = 0; i < 16; ++i) {
      const int ka = kk + (half << 4) + i;
      a[i]  = (_Float16)aw[(size_t)(l0 + mr) * LLN + ka];
      const int kb = kk + (half << 4) + i;
      bm[i] = (_Float16)q[(size_t)kb * DDM + n0 + mr];
    }
    c = wmma_f16(a, bm, c);
  }
  #pragma unroll
  for (int i = 0; i < 8; ++i)
    attn[((size_t)b * LLN + l0 + i + 8 * half) * DDM + h * DHH + n0 + mr] = c[i];
}

// ---------------------------------------------------------------------------
// Kernel 6/11: out = LayerNorm(a + b) * gamma + beta     (one block per row)
// ---------------------------------------------------------------------------
__global__ void k_add_ln(const float* __restrict__ a, const float* __restrict__ r,
                         const float* __restrict__ gamma,
                         const float* __restrict__ beta,
                         float* __restrict__ out) {
  __shared__ float red[256];
  const int row = blockIdx.x, tid = threadIdx.x;
  const float v0 = a[(size_t)row * DDM + tid] + r[(size_t)row * DDM + tid];
  const float v1 = a[(size_t)row * DDM + tid + 256] + r[(size_t)row * DDM + tid + 256];
  red[tid] = v0 + v1; __syncthreads();
  for (int s = 128; s > 0; s >>= 1) {
    if (tid < s) red[tid] += red[tid + s];
    __syncthreads();
  }
  const float mu = red[0] * (1.0f / DDM); __syncthreads();
  const float d0 = v0 - mu, d1 = v1 - mu;
  red[tid] = d0 * d0 + d1 * d1; __syncthreads();
  for (int s = 128; s > 0; s >>= 1) {
    if (tid < s) red[tid] += red[tid + s];
    __syncthreads();
  }
  const float rs = rsqrtf(red[0] * (1.0f / DDM) + 1e-5f);
  out[(size_t)row * DDM + tid]       = d0 * rs * gamma[tid] + beta[tid];
  out[(size_t)row * DDM + tid + 256] = d1 * rs * gamma[tid + 256] + beta[tid + 256];
}

// ---------------------------------------------------------------------------
// Kernel 7: transpose + f32->f16:  wt[c][r] = (f16) w[r][c]   (LDS tiled)
// block (32,8), grid (cols/32, rows/32)
// ---------------------------------------------------------------------------
__global__ void k_transpose_f16(const float* __restrict__ w,
                                _Float16* __restrict__ wt,
                                int rows, int cols) {
  __shared__ _Float16 tile[32][33];
  const int c0 = blockIdx.x * 32, r0 = blockIdx.y * 32;
  const int tx = threadIdx.x, ty = threadIdx.y;
  #pragma unroll
  for (int i = 0; i < 32; i += 8)
    tile[ty + i][tx] = (_Float16)w[(size_t)(r0 + ty + i) * cols + c0 + tx];
  __syncthreads();
  #pragma unroll
  for (int i = 0; i < 32; i += 8)
    wt[(size_t)(c0 + ty + i) * rows + r0 + tx] = tile[tx][ty + i];
}

// ---------------------------------------------------------------------------
// Kernel 8: f32 -> f16 copy (for the FFN A operand)
// ---------------------------------------------------------------------------
__global__ void k_cvt_f16(const float* __restrict__ x, _Float16* __restrict__ xh) {
  const int i = blockIdx.x * 1024 + threadIdx.x;
  #pragma unroll
  for (int j = 0; j < 4; ++j) xh[i + j * 256] = (_Float16)x[i + j * 256];
}

// ---------------------------------------------------------------------------
// Kernel 9 (WMMA): hbuf(f16) = leaky_relu(xh @ wt1^T + b1)  M=512 N=2048 K=512
// A,B pre-converted to f16; fragments are contiguous vector loads
// ---------------------------------------------------------------------------
__global__ void k_ffn1(const _Float16* __restrict__ xh,
                       const _Float16* __restrict__ wt1,   // [N=2048][K=512]
                       const float* __restrict__ b1, _Float16* __restrict__ hbuf) {
  const int m0 = blockIdx.x * 16, n0 = blockIdx.y * 64;
  const int lane = threadIdx.x, mr = lane & 15, half = lane >> 4;
  v8f c[4] = {};
  #pragma unroll 1
  for (int kk = 0; kk < DDM; kk += 32) {
    if (kk + 32 < DDM)
      __builtin_prefetch(&wt1[(size_t)(n0 + mr) * DDM + kk + 32], 0, 1);
    const _Float16* arow = xh + (size_t)(m0 + mr) * DDM + kk;
    const v8h alo = *(const v8h*)(arow + 16 * half);
    const v8h ahi = *(const v8h*)(arow + 16 * half + 8);
    v16h a;
    #pragma unroll
    for (int i = 0; i < 8; ++i) { a[i] = alo[i]; a[8 + i] = ahi[i]; }
    #pragma unroll
    for (int nt = 0; nt < 4; ++nt) {
      const v16h bm = *(const v16h*)(wt1 + (size_t)(n0 + nt * 16 + mr) * DDM + kk + 16 * half);
      c[nt] = wmma_f16(a, bm, c[nt]);
    }
  }
  #pragma unroll
  for (int nt = 0; nt < 4; ++nt)
    #pragma unroll
    for (int i = 0; i < 8; ++i) {
      const int col = n0 + nt * 16 + mr;
      float v = c[nt][i] + b1[col];
      v = v > 0.f ? v : 0.01f * v;
      hbuf[(size_t)(m0 + i + 8 * half) * DFF + col] = (_Float16)v;
    }
}

// ---------------------------------------------------------------------------
// Kernel 10 (WMMA): ybuf(f32) = hbuf @ wt2^T + b2   M=512 N=512 K=2048
// ---------------------------------------------------------------------------
__global__ void k_ffn2(const _Float16* __restrict__ hbuf,
                       const _Float16* __restrict__ wt2,   // [N=512][K=2048]
                       const float* __restrict__ b2, float* __restrict__ ybuf) {
  const int m0 = blockIdx.x * 16, n0 = blockIdx.y * 64;
  const int lane = threadIdx.x, mr = lane & 15, half = lane >> 4;
  v8f c[4] = {};
  #pragma unroll 1
  for (int kk = 0; kk < DFF; kk += 32) {
    if (kk + 32 < DFF)
      __builtin_prefetch(&wt2[(size_t)(n0 + mr) * DFF + kk + 32], 0, 1);
    const _Float16* arow = hbuf + (size_t)(m0 + mr) * DFF + kk;
    const v8h alo = *(const v8h*)(arow + 16 * half);
    const v8h ahi = *(const v8h*)(arow + 16 * half + 8);
    v16h a;
    #pragma unroll
    for (int i = 0; i < 8; ++i) { a[i] = alo[i]; a[8 + i] = ahi[i]; }
    #pragma unroll
    for (int nt = 0; nt < 4; ++nt) {
      const v16h bm = *(const v16h*)(wt2 + (size_t)(n0 + nt * 16 + mr) * DFF + kk + 16 * half);
      c[nt] = wmma_f16(a, bm, c[nt]);
    }
  }
  #pragma unroll
  for (int nt = 0; nt < 4; ++nt)
    #pragma unroll
    for (int i = 0; i < 8; ++i) {
      const int col = n0 + nt * 16 + mr;
      ybuf[(size_t)(m0 + i + 8 * half) * DDM + col] = c[nt][i] + b2[col];
    }
}

// ---------------------------------------------------------------------------
extern "C" void kernel_launch(void* const* d_in, const int* in_sizes, int n_in,
                              void* d_out, int out_size, void* d_ws, size_t ws_size,
                              hipStream_t stream) {
  (void)in_sizes; (void)n_in; (void)out_size; (void)ws_size;
  const float* src      = (const float*)d_in[0];
  const float* rel_diss = (const float*)d_in[1];
  const float* rel_dirs = (const float*)d_in[2];
  const float* rp_w     = (const float*)d_in[3];
  // d_in[4] = rp_b: constant over the softmax axis -> provably a no-op, skipped.
  const float* w1  = (const float*)d_in[5];
  const float* b1  = (const float*)d_in[6];
  const float* w2  = (const float*)d_in[7];
  const float* b2  = (const float*)d_in[8];
  const float* g1  = (const float*)d_in[9];
  const float* be1 = (const float*)d_in[10];
  const float* g2  = (const float*)d_in[11];
  const float* be2 = (const float*)d_in[12];
  float* out = (float*)d_out;

  char* p = (char*)d_ws;
  auto carve = [&](size_t bytes) {
    char* r = p;
    p += (bytes + 255) & ~(size_t)255;
    return r;
  };
  float*     t      = (float*)    carve((size_t)BB * LLN * HH * FFE * 4);  // 9.4 MB
  float*     scores = (float*)    carve((size_t)BB * HH * LLN * LLN * 4);  // 4.2 MB
  float*     attn   = (float*)    carve((size_t)BB * LLN * DDM * 4);       // 1.0 MB
  float*     xbuf   = (float*)    carve((size_t)BB * LLN * DDM * 4);       // 1.0 MB
  _Float16*  xh     = (_Float16*) carve((size_t)BB * LLN * DDM * 2);       // 0.5 MB
  _Float16*  wt1    = (_Float16*) carve((size_t)DFF * DDM * 2);            // 2.1 MB
  _Float16*  wt2    = (_Float16*) carve((size_t)DDM * DFF * 2);            // 2.1 MB
  _Float16*  hbuf   = (_Float16*) carve((size_t)BB * LLN * DFF * 2);       // 2.1 MB
  float*     ybuf   = (float*)    carve((size_t)BB * LLN * DDM * 4);       // 1.0 MB

  // Weight prep (independent of activations)
  k_transpose_f16<<<dim3(DFF / 32, DDM / 32), dim3(32, 8), 0, stream>>>(w1, wt1, DDM, DFF);
  k_transpose_f16<<<dim3(DDM / 32, DFF / 32), dim3(32, 8), 0, stream>>>(w2, wt2, DFF, DDM);

  // Attention path
  k_rpproj_t   <<<BB * LLN, 256, 0, stream>>>(src, rp_w, t);
  k_base_scores<<<dim3(LLN / 16, LLN / 16, BB * HH), 32, 0, stream>>>(src, scores);
  k_rp_scores  <<<BB * LLN, 256, 0, stream>>>(rel_diss, rel_dirs, t, scores);
  k_softmax    <<<BB * HH * LLN, 256, 0, stream>>>(scores);
  k_attn_v     <<<dim3(LLN / 16, DHH / 16, BB * HH), 32, 0, stream>>>(scores, src, attn);
  k_add_ln     <<<BB * LLN, 256, 0, stream>>>(src, attn, g1, be1, xbuf);

  // FFN path (f16 operands, f32 accumulate)
  k_cvt_f16    <<<(BB * LLN * DDM) / 1024, 256, 0, stream>>>(xbuf, xh);
  k_ffn1       <<<dim3((BB * LLN) / 16, DFF / 64), 32, 0, stream>>>(xh, wt1, b1, hbuf);
  k_ffn2       <<<dim3((BB * LLN) / 16, DDM / 64), 32, 0, stream>>>(hbuf, wt2, b2, ybuf);
  k_add_ln     <<<BB * LLN, 256, 0, stream>>>(xbuf, ybuf, g2, be2, out);
}
